// SimpleRNN_83227876262031
// MI455X (gfx1250) — matrix operand order repaired and checked by reading, not compiled, not run on hardware
//
#include <hip/hip_runtime.h>
#include <hip/hip_bf16.h>
#include <stdint.h>

// Problem dims (from reference)
#define B_TOT   256
#define T_LEN   1024
#define E_DIM   100     // embedding width (zero-padded to 128 for K)
#define H_DIM   128
#define K_PAD   128
#define KP      (K_PAD + 8)   // +8 halfwords row padding: stride 272B, kills bank conflicts
#define HID     50
#define OUT_DIM 2
#define ROWS    16      // batch rows per block (one 16-row M tile)

typedef __attribute__((ext_vector_type(16))) __bf16 v16bf;
typedef __attribute__((ext_vector_type(8)))  float  v8f;

union Frag { uint32_t u[8]; v16bf v; };

// float -> bf16 bits, round-to-nearest-even (fallback path)
__device__ __forceinline__ uint16_t f2bf(float f) {
  uint32_t u = __builtin_bit_cast(uint32_t, f);
  return (uint16_t)((u + 0x7FFFu + ((u >> 16) & 1u)) >> 16);
}

// pack two floats into packed bf16 (prefer hardware v_cvt_pk_bf16_f32)
__device__ __forceinline__ uint32_t pack_bf16(float lo, float hi) {
#if __has_builtin(__builtin_amdgcn_cvt_pk_bf16_f32)
  auto p = __builtin_amdgcn_cvt_pk_bf16_f32(lo, hi);
  return __builtin_bit_cast(uint32_t, p);
#else
  return (uint32_t)f2bf(lo) | ((uint32_t)f2bf(hi) << 16);
#endif
}

// hardware transcendental tanh (v_tanh_f32 on gfx1250)
__device__ __forceinline__ float fast_tanh(float x) {
#if __has_builtin(__builtin_amdgcn_tanhf)
  return __builtin_amdgcn_tanhf(x);
#else
  float r;
  asm volatile("v_tanh_f32 %0, %1\n\tv_nop" : "=v"(r) : "v"(x));
  return r;
#endif
}

// ---------------------------------------------------------------------------
// Fused embedding-gather + input-projection + masked RNN scan.
// grid = 16 blocks (16 batch rows each), block = 256 threads = 8 wave32s.
// Wave w owns output columns [16w, 16w+16). Writes final h state [B,H] to ws.
// One barrier per step (sH double-buffered); per-step critical path is
// 16 ds_load_b128 -> 8 chained v_wmma_f32_16x16x32_bf16 -> 8 v_tanh_f32.
// ---------------------------------------------------------------------------
__global__ __launch_bounds__(256)
void rnn_scan_kernel(const int* __restrict__ ids,
                     const float* __restrict__ embW,
                     const float* __restrict__ Wx,
                     const float* __restrict__ Wh,
                     const float* __restrict__ bias,
                     float* __restrict__ hT)
{
  __shared__ __attribute__((aligned(16))) uint16_t sWT[H_DIM][KP];     // 34KB weight staging
  __shared__ __attribute__((aligned(16))) uint16_t sEmb[2][ROWS][KP];  // 8.5KB dbl-buf emb tile
  __shared__ __attribute__((aligned(16))) uint16_t sH[2][ROWS][KP];    // 8.5KB dbl-buf hidden
  __shared__ uint32_t sMaskW[T_LEN];                                   // 4KB: bit r = mask(row r, t)

  const int tid   = threadIdx.x;
  const int wave  = tid >> 5;
  const int lane  = tid & 31;
  const int nloc  = lane & 15;     // N within 16-wide tile / A-matrix row M
  const int khalf = lane >> 4;     // lane-group select in WMMA layouts
  const int b0    = blockIdx.x * ROWS;

  // ---- Mask words: sMaskW[t] bit r = (ids[b0+r][t] != 0) ----
  for (int t = tid; t < T_LEN; t += 256) {
    uint32_t w = 0;
    #pragma unroll
    for (int r = 0; r < ROWS; ++r)
      w |= (ids[(size_t)(b0 + r) * T_LEN + t] != 0 ? 1u : 0u) << r;
    sMaskW[t] = w;
  }

  // ---- Wh -> transposed bf16 staging -> per-wave B fragments (live in VGPRs) ----
  for (int i = tid; i < H_DIM * H_DIM; i += 256) {
    const int k = i >> 7, n = i & 127;           // Wh[k][n], coalesced global read
    sWT[n][k] = f2bf(Wh[i]);
  }
  __syncthreads();
  Frag whB[4];
  #pragma unroll
  for (int c = 0; c < 4; ++c)
    #pragma unroll
    for (int v = 0; v < 8; ++v) {
      // B 32x16 bf16, contiguous k order used here (A matches): lane = N + 16*(K/16); VGPR v holds K pair (2v,2v+1)
      const int n = (wave << 4) + nloc;
      const int k = (c << 5) + (khalf << 4) + (v << 1);
      whB[c].u[v] = *reinterpret_cast<const uint32_t*>(&sWT[n][k]);
    }
  __syncthreads();

  // ---- Wx (E=100 rows, zero-padded to 128) -> B fragments ----
  for (int i = tid; i < H_DIM * K_PAD; i += 256) {
    const int n = i >> 7, k = i & 127;
    sWT[n][k] = (k < E_DIM) ? f2bf(Wx[(size_t)k * H_DIM + n]) : (uint16_t)0;
  }
  __syncthreads();
  Frag wxB[4];
  #pragma unroll
  for (int c = 0; c < 4; ++c)
    #pragma unroll
    for (int v = 0; v < 8; ++v) {
      const int n = (wave << 4) + nloc;
      const int k = (c << 5) + (khalf << 4) + (v << 1);
      wxB[c].u[v] = *reinterpret_cast<const uint32_t*>(&sWT[n][k]);
    }

  // ---- h(t=0 input) = 0 in buffer 0; stage embeddings for t=0 into buffer 0 ----
  for (int i = tid; i < ROWS * KP; i += 256) {
    const int row = i / KP, k = i % KP;
    sH[0][row][k] = 0;
  }
  for (int i = tid; i < ROWS * (K_PAD / 2); i += 256) {   // pair-granular staging
    const int row = i >> 6, kp = (i & 63) << 1;
    const int id = ids[(size_t)(b0 + row) * T_LEN];
    const float* er = embW + (size_t)id * E_DIM;
    const float lo = (kp < E_DIM) ? er[kp] : 0.f;
    const float hi = (kp + 1 < E_DIM) ? er[kp + 1] : 0.f;
    *reinterpret_cast<uint32_t*>(&sEmb[0][row][kp]) = pack_bf16(lo, hi);
  }

  const float bval = bias[(wave << 4) + nloc];  // bias at this lane's N column
  float hp[8];                                  // running h at this lane's 8 C/D slots
  #pragma unroll
  for (int r = 0; r < 8; ++r) hp[r] = 0.f;

  __syncthreads();

  for (int t = 0; t < T_LEN; ++t) {
    const int buf = t & 1, nbuf = buf ^ 1;

    // ---- Stage embeddings for t+1 ASAP (global loads overlap the WMMA chain) ----
    if (t + 1 < T_LEN) {
      for (int i = tid; i < ROWS * (K_PAD / 2); i += 256) {  // 4 iterations
        const int row = i >> 6, kp = (i & 63) << 1;
        const int id = ids[(size_t)(b0 + row) * T_LEN + (t + 1)];
        const float* er = embW + (size_t)id * E_DIM;
        const float lo = (kp < E_DIM) ? er[kp] : 0.f;
        const float hi = (kp + 1 < E_DIM) ? er[kp + 1] : 0.f;
        *reinterpret_cast<uint32_t*>(&sEmb[nbuf][row][kp]) = pack_bf16(lo, hi);
        if (kp == 0 && t + 2 < T_LEN) {                      // warm caches for t+2
          const int id2 = ids[(size_t)(b0 + row) * T_LEN + (t + 2)];
          __builtin_prefetch(&embW[(size_t)id2 * E_DIM], 0, 3);
        }
      }
    }

    // ---- Build A fragments from LDS (16-bit A 16x32 layout) ----
    Frag aE[4], aH[4];
    #pragma unroll
    for (int c = 0; c < 4; ++c)
      #pragma unroll
      for (int v = 0; v < 8; ++v) {
        // lane = M (nloc); K = 32c + 16*khalf + 2v
        const int k = (c << 5) + (khalf << 4) + (v << 1);
        aE[c].u[v] = *reinterpret_cast<const uint32_t*>(&sEmb[buf][nloc][k]);
        aH[c].u[v] = *reinterpret_cast<const uint32_t*>(&sH[buf][nloc][k]);
      }

    // ---- acc = b + emb@Wx + h@Wh  (8 chained v_wmma_f32_16x16x32_bf16) ----
    v8f acc = { bval, bval, bval, bval, bval, bval, bval, bval };
    #pragma unroll
    for (int c = 0; c < 4; ++c)
      acc = __builtin_amdgcn_wmma_f32_16x16x32_bf16(
          false, aE[c].v, false, wxB[c].v, (short)0, acc, false, false);
    #pragma unroll
    for (int c = 0; c < 4; ++c)
      acc = __builtin_amdgcn_wmma_f32_16x16x32_bf16(
          false, aH[c].v, false, whB[c].v, (short)0, acc, false, false);

    // ---- hardware tanh + Keras masking (id==0 keeps previous state) ----
    const uint32_t mw = sMaskW[t];                    // uniform: one LDS word per step
    #pragma unroll
    for (int r = 0; r < 8; ++r) {
      const int row = r + (khalf << 3);               // C/D layout: M = r + 8*(lane/16)
      const float hn = fast_tanh(acc[r]);
      hp[r] = ((mw >> row) & 1u) ? hn : hp[r];
    }

    // ---- publish h into the other buffer; single barrier per step ----
    #pragma unroll
    for (int r = 0; r < 8; ++r) {
      const int row = r + (khalf << 3);
      sH[nbuf][row][(wave << 4) + nloc] = f2bf(hp[r]);
    }
    __syncthreads();                                  // writes of sH[nbuf]/sEmb[nbuf] visible
  }

  // ---- Write final hidden state (f32, full precision from registers) ----
  #pragma unroll
  for (int r = 0; r < 8; ++r) {
    const int row = r + (khalf << 3);
    hT[(size_t)(b0 + row) * H_DIM + (wave << 4) + nloc] = hp[r];
  }
}

// ---------------------------------------------------------------------------
// Head: out = sigmoid(relu(hT@W1 + b1) @ W2 + b2). Tiny: one block, 1 row/thread.
// ---------------------------------------------------------------------------
__global__ __launch_bounds__(256)
void head_kernel(const float* __restrict__ hT,
                 const float* __restrict__ W1, const float* __restrict__ b1,
                 const float* __restrict__ W2, const float* __restrict__ b2,
                 float* __restrict__ out)
{
  __shared__ float sW1[H_DIM][HID];         // 25.6KB
  __shared__ float sW2[HID][OUT_DIM];
  __shared__ float sB1[HID];
  __shared__ float sB2[OUT_DIM];
  const int tid = threadIdx.x;
  for (int i = tid; i < H_DIM * HID; i += 256) sW1[i / HID][i % HID] = W1[i];
  for (int i = tid; i < HID * OUT_DIM; i += 256) (&sW2[0][0])[i] = W2[i];
  for (int i = tid; i < HID; i += 256) sB1[i] = b1[i];
  for (int i = tid; i < OUT_DIM; i += 256) sB2[i] = b2[i];
  __syncthreads();

  float a[HID];
  #pragma unroll
  for (int i = 0; i < HID; ++i) a[i] = sB1[i];

  const float* hrow = hT + (size_t)tid * H_DIM;
  for (int k = 0; k < H_DIM; ++k) {
    const float hv = hrow[k];
    #pragma unroll
    for (int i = 0; i < HID; ++i) a[i] += hv * sW1[k][i];
  }

  float o[OUT_DIM];
  #pragma unroll
  for (int j = 0; j < OUT_DIM; ++j) o[j] = sB2[j];
  #pragma unroll
  for (int i = 0; i < HID; ++i) {
    const float r = fmaxf(a[i], 0.f);
    #pragma unroll
    for (int j = 0; j < OUT_DIM; ++j) o[j] += r * sW2[i][j];
  }
  #pragma unroll
  for (int j = 0; j < OUT_DIM; ++j)
    out[(size_t)tid * OUT_DIM + j] = 1.f / (1.f + expf(-o[j]));
}

// ---------------------------------------------------------------------------
extern "C" void kernel_launch(void* const* d_in, const int* in_sizes, int n_in,
                              void* d_out, int out_size, void* d_ws, size_t ws_size,
                              hipStream_t stream) {
  const int*   ids  = (const int*)  d_in[0];
  const float* embW = (const float*)d_in[1];
  const float* Wx   = (const float*)d_in[2];
  const float* Wh   = (const float*)d_in[3];
  const float* b    = (const float*)d_in[4];
  const float* W1   = (const float*)d_in[5];
  const float* b1   = (const float*)d_in[6];
  const float* W2   = (const float*)d_in[7];
  const float* b2   = (const float*)d_in[8];

  float* hT = (float*)d_ws;   // needs B_TOT*H_DIM*4 = 128KB of workspace

  rnn_scan_kernel<<<B_TOT / ROWS, 256, 0, stream>>>(ids, embW, Wx, Wh, b, hT);
  head_kernel<<<1, 256, 0, stream>>>(hT, W1, b1, W2, b2, (float*)d_out);
}
